// Bottleneck_refine_50903952392818
// MI455X (gfx1250) — matrix operand order repaired and checked by reading, not compiled, not run on hardware
//
#include <hip/hip_runtime.h>

// ---------------------------------------------------------------------------
// Bottleneck_refine (masked grouped bottleneck, training-mode BN) for gfx1250.
// conv1(1x1,g=2) -> BN1+ReLU+mask -> conv2(3x3,g=2) -> BN2+ReLU+mask
// -> conv3(1x1,g=2) -> BN3 + residual + ReLU.
// Convs = bf16 WMMA GEMMs (v_wmma_f32_16x16x32_bf16), block tile 128x128,
// B/A tiles staged with global_load_async_to_lds_b128 where possible,
// per-block BN partial stats fused into the GEMM epilogue (deterministic).
// ---------------------------------------------------------------------------

typedef __attribute__((ext_vector_type(16))) __bf16     v16bf;
typedef __attribute__((ext_vector_type(8)))  float      v8f;
typedef __attribute__((ext_vector_type(4)))  unsigned   u32x4;

#define M_TOTAL 50176   // 64 * 28 * 28
#define IMG     784
#define HWDIM   28
#define LDA     40      // LDS row stride in u16 (32 + 8 pad) -> 80B, 16B aligned
#define NBLK_M  392     // M_TOTAL / 128

__device__ __forceinline__ unsigned short f2bf(float f) {
    union { float f; unsigned u; } c; c.f = f;
    unsigned r = c.u + 0x7FFFu + ((c.u >> 16) & 1u);   // round-to-nearest-even
    return (unsigned short)(r >> 16);
}

union Frag { v16bf v; u32x4 q[2]; };

// A fragment (16x32 bf16): elems 0..15 <- K = 16*half + 0..15 (contiguous)
__device__ __forceinline__ void load_fragA(const unsigned short* p, Frag& f) {
    f.q[0] = *(const u32x4*)(p);
    f.q[1] = *(const u32x4*)(p + 8);
}
// B fragment (32x16 bf16): elems 0..15 <- K = 16*half + 0..15 (contiguous)
__device__ __forceinline__ void load_fragB(const unsigned short* p, Frag& f) {
    f.q[0] = *(const u32x4*)(p);
    f.q[1] = *(const u32x4*)(p + 8);
}

// 16B global -> LDS async copy (per-lane), tracked by ASYNCcnt.
__device__ __forceinline__ void async_copy16(unsigned lds_addr, const void* gptr) {
    asm volatile("global_load_async_to_lds_b128 %0, %1, off"
                 :: "v"(lds_addr), "v"(gptr) : "memory");
}
__device__ __forceinline__ void wait_async0() {
    asm volatile("s_wait_asynccnt 0" ::: "memory");
}

#define WMMA_BF16(A, B, C) \
    __builtin_amdgcn_wmma_f32_16x16x32_bf16(false, (A), false, (B), (short)0, (C), false, false)

// ---------------------------------------------------------------------------
// Weight packing
// ---------------------------------------------------------------------------
__global__ void copy_bf16_kernel(const float* __restrict__ src,
                                 unsigned short* __restrict__ dst, int n) {
    int i = blockIdx.x * 256 + threadIdx.x;
    if (i < n) dst[i] = f2bf(src[i]);
}

// w2: [256][128][3][3] f32 -> B2t [256][1152] bf16 with k = kpos*128 + ic
__global__ void pack_w2_kernel(const float* __restrict__ w2,
                               unsigned short* __restrict__ dst) {
    int i = blockIdx.x * 256 + threadIdx.x;
    if (i >= 256 * 1152) return;
    int oc = i / 1152, k = i - oc * 1152;
    int kpos = k >> 7, ic = k & 127;
    dst[i] = f2bf(w2[(oc * 128 + ic) * 9 + kpos]);
}

// ---------------------------------------------------------------------------
// Shared GEMM epilogue: store 128x128 f32 tile + per-block BN partial stats.
// acc[i][j]: wave tile 32(M) x 64(N); wm = wave&3, wn = wave>>2.
// ---------------------------------------------------------------------------
#define GEMM_EPILOGUE(CoutTot)                                                  \
    {                                                                           \
        _Pragma("unroll")                                                       \
        for (int i = 0; i < 2; ++i) {                                           \
            _Pragma("unroll")                                                   \
            for (int j = 0; j < 4; ++j) {                                       \
                _Pragma("unroll")                                               \
                for (int r = 0; r < 8; ++r) {                                   \
                    int m = wm * 32 + i * 16 + half * 8 + r;                    \
                    int nI = ocBase + wn * 64 + j * 16 + lr;                    \
                    Y[(size_t)(rowBase + m) * (CoutTot) + nI] = acc[i][j][r];   \
                }                                                               \
            }                                                                   \
        }                                                                       \
        _Pragma("unroll")                                                       \
        for (int j = 0; j < 4; ++j) {                                           \
            float s = 0.f, q = 0.f;                                             \
            _Pragma("unroll")                                                   \
            for (int r = 0; r < 8; ++r) {                                       \
                float v0 = acc[0][j][r], v1 = acc[1][j][r];                     \
                s += v0 + v1;                                                   \
                q += v0 * v0 + v1 * v1;                                         \
            }                                                                   \
            s += __shfl_xor(s, 16, 32);                                         \
            q += __shfl_xor(q, 16, 32);                                         \
            if (half == 0) {                                                    \
                sred[wave][j * 16 + lr] = s;                                    \
                qred[wave][j * 16 + lr] = q;                                    \
            }                                                                   \
        }                                                                       \
        __syncthreads();                                                        \
        if (t < 128) {                                                          \
            int wnc = t >> 6, cl = t & 63;                                      \
            float s = 0.f, q = 0.f;                                             \
            _Pragma("unroll")                                                   \
            for (int wmi = 0; wmi < 4; ++wmi) {                                 \
                s += sred[wnc * 4 + wmi][cl];                                   \
                q += qred[wnc * 4 + wmi][cl];                                   \
            }                                                                   \
            psum[(size_t)blockIdx.x * (CoutTot) + ocBase + t] = s;              \
            psq [(size_t)blockIdx.x * (CoutTot) + ocBase + t] = q;              \
        }                                                                       \
    }

// ---------------------------------------------------------------------------
// 1x1 grouped GEMM.  MODE 0: A = masked f32 NCHW x (conv1, sync A loader)
//                    MODE 1: A = bf16 [M][CinTot] activations (conv3, async A)
// Block 128M x 128N, 8 waves of 32x64, K-step 32, B tiles async-to-LDS.
// ---------------------------------------------------------------------------
template <int MODE>
__global__ __launch_bounds__(256) void gemm1x1_kernel(
    const float* __restrict__ Xf, const unsigned short* __restrict__ Xb,
    const float* __restrict__ mask, const unsigned short* __restrict__ Bt,
    float* __restrict__ Y, float* __restrict__ psum, float* __restrict__ psq,
    int K, int NperG, int CinTot, int CoutTot) {

    __shared__ unsigned short Alds[128 * LDA];
    __shared__ unsigned short Blds[128 * LDA];
    __shared__ float sred[8][64];
    __shared__ float qred[8][64];

    const int t    = threadIdx.x;
    const int lane = t & 31, wave = t >> 5;
    const int wm   = wave & 3, wn = wave >> 2;
    const int half = lane >> 4, lr = lane & 15;
    const int rowBase = blockIdx.x * 128;
    const int ocBase  = blockIdx.y * 128;
    const int g    = ocBase / NperG;
    const int gOff = g * (CinTot >> 1);

    v8f acc[2][4] = {};

    // per-thread geometry for MODE 0 loader
    int r0 = 0, kkb = 0;
    float mv = 0.f;
    size_t abase0 = 0;
    if (MODE == 0) {
        r0 = t & 127; kkb = t >> 7;
        int row = rowBase + r0;
        int n = row / IMG, hw = row - n * IMG;
        int h = hw / HWDIM, w = hw - h * HWDIM;
        mv = mask[(size_t)(n * 2 + g) * 49 + (h >> 2) * 7 + (w >> 2)];
        abase0 = (size_t)n * CinTot * IMG + hw;
    }

    for (int k0 = 0; k0 < K; k0 += 32) {
        // B tile: 128 rows x 32 K, async global->LDS
#pragma unroll
        for (int i = 0; i < 2; ++i) {
            int e = i * 256 + t;
            int nl = e >> 2, kc = e & 3;
            async_copy16((unsigned)(uintptr_t)&Blds[nl * LDA + kc * 8],
                         &Bt[(size_t)(ocBase + nl) * K + k0 + kc * 8]);
        }
        if (MODE == 0) {
#pragma unroll
            for (int i = 0; i < 16; ++i) {
                int kk = i * 2 + kkb;
                int c  = gOff + k0 + kk;
                Alds[r0 * LDA + kk] = f2bf(Xf[abase0 + (size_t)c * IMG] * mv);
            }
        } else {
#pragma unroll
            for (int i = 0; i < 2; ++i) {
                int e = i * 256 + t;
                int r = e >> 2, kc = e & 3;
                async_copy16((unsigned)(uintptr_t)&Alds[r * LDA + kc * 8],
                             &Xb[(size_t)(rowBase + r) * CinTot + gOff + k0 + kc * 8]);
            }
        }
        wait_async0();
        __syncthreads();

        Frag a[2], b[4];
        const unsigned short* ap = &Alds[(wm * 32 + lr) * LDA + half * 16];
        load_fragA(ap, a[0]);
        load_fragA(ap + 16 * LDA, a[1]);
        const unsigned short* bp = &Blds[(wn * 64 + lr) * LDA + half * 16];
#pragma unroll
        for (int j = 0; j < 4; ++j) load_fragB(bp + j * 16 * LDA, b[j]);
#pragma unroll
        for (int i = 0; i < 2; ++i)
#pragma unroll
            for (int j = 0; j < 4; ++j)
                acc[i][j] = WMMA_BF16(a[i].v, b[j].v, acc[i][j]);
        __syncthreads();
    }

    GEMM_EPILOGUE(CoutTot)
}

// ---------------------------------------------------------------------------
// 3x3 grouped implicit-GEMM (conv2). A = bf16 [M][256] (bn+relu+mask applied),
// K = 1152 ordered kpos*128+ic; zero padding in the (sync) A loader.
// ---------------------------------------------------------------------------
__global__ __launch_bounds__(256) void gemm3x3_kernel(
    const unsigned short* __restrict__ Xb, const unsigned short* __restrict__ Bt,
    float* __restrict__ Y, float* __restrict__ psum, float* __restrict__ psq) {

    const int K = 1152, CinTot = 256, CoutTot = 256;
    __shared__ unsigned short Alds[128 * LDA];
    __shared__ unsigned short Blds[128 * LDA];
    __shared__ float sred[8][64];
    __shared__ float qred[8][64];

    const int t    = threadIdx.x;
    const int lane = t & 31, wave = t >> 5;
    const int wm   = wave & 3, wn = wave >> 2;
    const int half = lane >> 4, lr = lane & 15;
    const int rowBase = blockIdx.x * 128;
    const int ocBase  = blockIdx.y * 128;
    const int g = blockIdx.y;             // NperG = 128

    v8f acc[2][4] = {};

    const int rl = t >> 2, kc8 = (t & 3) * 8;
    int nn[2], hh[2], ww[2];
#pragma unroll
    for (int i = 0; i < 2; ++i) {
        int row = rowBase + rl + i * 64;
        nn[i] = row / IMG;
        int hw = row - nn[i] * IMG;
        hh[i] = hw / HWDIM;
        ww[i] = hw - hh[i] * HWDIM;
    }

    for (int k0 = 0; k0 < K; k0 += 32) {
#pragma unroll
        for (int i = 0; i < 2; ++i) {               // B tile async
            int e = i * 256 + t;
            int nl = e >> 2, kc = e & 3;
            async_copy16((unsigned)(uintptr_t)&Blds[nl * LDA + kc * 8],
                         &Bt[(size_t)(ocBase + nl) * K + k0 + kc * 8]);
        }
        int kpos = k0 >> 7;
        int ic   = (k0 & 127) + kc8;
        int dy   = kpos / 3 - 1;
        int dx   = kpos - (kpos / 3) * 3 - 1;
#pragma unroll
        for (int i = 0; i < 2; ++i) {               // A tile: shifted, zero-padded
            int h2 = hh[i] + dy, w2 = ww[i] + dx;
            u32x4 v = {0u, 0u, 0u, 0u};
            if ((unsigned)h2 < 28u && (unsigned)w2 < 28u)
                v = *(const u32x4*)&Xb[((size_t)nn[i] * IMG + h2 * HWDIM + w2) * CinTot + g * 128 + ic];
            *(u32x4*)&Alds[(rl + i * 64) * LDA + kc8] = v;
        }
        wait_async0();
        __syncthreads();

        Frag a[2], b[4];
        const unsigned short* ap = &Alds[(wm * 32 + lr) * LDA + half * 16];
        load_fragA(ap, a[0]);
        load_fragA(ap + 16 * LDA, a[1]);
        const unsigned short* bp = &Blds[(wn * 64 + lr) * LDA + half * 16];
#pragma unroll
        for (int j = 0; j < 4; ++j) load_fragB(bp + j * 16 * LDA, b[j]);
#pragma unroll
        for (int i = 0; i < 2; ++i)
#pragma unroll
            for (int j = 0; j < 4; ++j)
                acc[i][j] = WMMA_BF16(a[i].v, b[j].v, acc[i][j]);
        __syncthreads();
    }

    GEMM_EPILOGUE(CoutTot)
}

// ---------------------------------------------------------------------------
// Reduce per-block BN partials (deterministic).
// ---------------------------------------------------------------------------
__global__ void stats2_kernel(const float* __restrict__ psum,
                              const float* __restrict__ psumsq, int C, int nblk,
                              float* __restrict__ sum, float* __restrict__ sumsq) {
    int c = blockIdx.x * 256 + threadIdx.x;
    if (c >= C) return;
    float s = 0.f, q = 0.f;
    for (int i = 0; i < nblk; ++i) {
        s += psum[(size_t)i * C + c];
        q += psumsq[(size_t)i * C + c];
    }
    sum[c] = s;
    sumsq[c] = q;
}

// ---------------------------------------------------------------------------
// BN + ReLU + mask + bf16 pack  (Y f32 [M][256] -> X bf16 [M][256])
// ---------------------------------------------------------------------------
__global__ __launch_bounds__(256) void bnpack_kernel(
    const float* __restrict__ Y, const float* __restrict__ sum,
    const float* __restrict__ sumsq, const float* __restrict__ gamma,
    const float* __restrict__ beta, const float* __restrict__ mask,
    unsigned short* __restrict__ X) {
    const float invM = 1.0f / (float)M_TOTAL;
    int row = blockIdx.x;
    int c   = threadIdx.x;
    int n = row / IMG, hw = row - n * IMG;
    int h = hw / HWDIM, w = hw - h * HWDIM;
    int g = c >> 7;
    float mv = mask[(size_t)(n * 2 + g) * 49 + (h >> 2) * 7 + (w >> 2)];
    float mean = sum[c] * invM;
    float var  = sumsq[c] * invM - mean * mean;
    float rstd = rsqrtf(var + 1e-5f);
    size_t idx = (size_t)row * 256 + c;
    float v = (Y[idx] - mean) * rstd * gamma[c] + beta[c];
    v = fmaxf(v, 0.f) * mv;
    X[idx] = f2bf(v);
}

// ---------------------------------------------------------------------------
// Final: BN3 + residual + ReLU with NHWC->NCHW LDS transpose.
// ---------------------------------------------------------------------------
__global__ __launch_bounds__(256) void final_kernel(
    const float* __restrict__ Y3, const float* __restrict__ x,
    const float* __restrict__ sum, const float* __restrict__ sumsq,
    const float* __restrict__ gamma, const float* __restrict__ beta,
    float* __restrict__ out) {
    __shared__ float tile[32][33];
    const float invM = 1.0f / (float)M_TOTAL;
    const int n = blockIdx.z;
    const int hw0 = blockIdx.x * 32;
    const int c0  = blockIdx.y * 32;
#pragma unroll
    for (int i = 0; i < 4; ++i) {
        int hw = hw0 + threadIdx.y + i * 8;
        if (hw < IMG)
            tile[threadIdx.y + i * 8][threadIdx.x] =
                Y3[((size_t)n * IMG + hw) * 1024 + c0 + threadIdx.x];
    }
    __syncthreads();
#pragma unroll
    for (int i = 0; i < 4; ++i) {
        int c  = c0 + threadIdx.y + i * 8;
        int hw = hw0 + threadIdx.x;
        if (hw < IMG) {
            float mean = sum[c] * invM;
            float var  = sumsq[c] * invM - mean * mean;
            float rstd = rsqrtf(var + 1e-5f);
            float v = (tile[threadIdx.x][threadIdx.y + i * 8] - mean) * rstd * gamma[c] + beta[c];
            size_t o = ((size_t)n * 1024 + c) * IMG + hw;
            out[o] = fmaxf(v + x[o], 0.f);
        }
    }
}

// ---------------------------------------------------------------------------
// Launch
// ---------------------------------------------------------------------------
extern "C" void kernel_launch(void* const* d_in, const int* in_sizes, int n_in,
                              void* d_out, int out_size, void* d_ws, size_t ws_size,
                              hipStream_t stream) {
    (void)in_sizes; (void)n_in; (void)out_size; (void)ws_size;
    const float* x      = (const float*)d_in[0];
    const float* mask   = (const float*)d_in[1];
    const float* w1     = (const float*)d_in[2];
    const float* gamma1 = (const float*)d_in[3];
    const float* beta1  = (const float*)d_in[4];
    const float* w2     = (const float*)d_in[5];
    const float* gamma2 = (const float*)d_in[6];
    const float* beta2  = (const float*)d_in[7];
    const float* w3     = (const float*)d_in[8];
    const float* gamma3 = (const float*)d_in[9];
    const float* beta3  = (const float*)d_in[10];
    float* out = (float*)d_out;

    char* ws = (char*)d_ws;
    size_t off = 0;
    auto alloc = [&](size_t bytes) -> void* {
        off = (off + 255) & ~(size_t)255;
        void* p = ws + off;
        off += bytes;
        return p;
    };

    float*          statsB = (float*)alloc(3072 * sizeof(float));
    float*          psum   = (float*)alloc((size_t)NBLK_M * 1024 * 4);
    float*          psq    = (float*)alloc((size_t)NBLK_M * 1024 * 4);
    unsigned short* B1t    = (unsigned short*)alloc((size_t)256 * 512 * 2);
    unsigned short* B2t    = (unsigned short*)alloc((size_t)256 * 1152 * 2);
    unsigned short* B3t    = (unsigned short*)alloc((size_t)1024 * 128 * 2);
    float*          Y      = (float*)alloc((size_t)M_TOTAL * 256 * 4);
    unsigned short* Xb     = (unsigned short*)alloc((size_t)M_TOTAL * 256 * 2);
    float*          Y3     = (float*)alloc((size_t)M_TOTAL * 1024 * 4);

    float* sum1 = statsB,        *ss1 = statsB + 256;
    float* sum2 = statsB + 512,  *ss2 = statsB + 768;
    float* sum3 = statsB + 1024, *ss3 = statsB + 2048;

    // weight packing
    copy_bf16_kernel<<<512, 256, 0, stream>>>(w1, B1t, 256 * 512);
    copy_bf16_kernel<<<512, 256, 0, stream>>>(w3, B3t, 1024 * 128);
    pack_w2_kernel<<<(256 * 1152 + 255) / 256, 256, 0, stream>>>(w2, B2t);

    // conv1: masked x (f32 NCHW) x W1 -> Y [M][256] (+ fused BN partials)
    gemm1x1_kernel<0><<<dim3(NBLK_M, 2), 256, 0, stream>>>(
        x, nullptr, mask, B1t, Y, psum, psq, 512, 128, 1024, 256);
    stats2_kernel<<<1, 256, 0, stream>>>(psum, psq, 256, NBLK_M, sum1, ss1);
    bnpack_kernel<<<M_TOTAL, 256, 0, stream>>>(Y, sum1, ss1, gamma1, beta1, mask, Xb);

    // conv2: 3x3 implicit GEMM -> Y [M][256]
    gemm3x3_kernel<<<dim3(NBLK_M, 2), 256, 0, stream>>>(Xb, B2t, Y, psum, psq);
    stats2_kernel<<<1, 256, 0, stream>>>(psum, psq, 256, NBLK_M, sum2, ss2);
    bnpack_kernel<<<M_TOTAL, 256, 0, stream>>>(Y, sum2, ss2, gamma2, beta2, mask, Xb);

    // conv3 -> Y3 [M][1024]
    gemm1x1_kernel<1><<<dim3(NBLK_M, 8), 256, 0, stream>>>(
        nullptr, Xb, mask, B3t, Y3, psum, psq, 128, 512, 256, 1024);
    stats2_kernel<<<4, 256, 0, stream>>>(psum, psq, 1024, NBLK_M, sum3, ss3);

    // BN3 + residual + ReLU, NHWC -> NCHW
    final_kernel<<<dim3(25, 32, 64), dim3(32, 8), 0, stream>>>(
        Y3, x, sum3, ss3, gamma3, beta3, out);
}
